// FSULinear_21457656610816
// MI455X (gfx1250) — 8-bit matrix operand order repaired and checked by reading, compiled, not run on hardware
//
#include <hip/hip_runtime.h>
#include <stdint.h>


typedef int v8i __attribute__((ext_vector_type(8)));
typedef int g4i __attribute__((vector_size(16)));
typedef __attribute__((address_space(1))) g4i g4i_as1;
typedef __attribute__((address_space(3))) g4i g4i_as3;

#define FSU_DEPTH 12
#define BLOCK_M   128
#define BLOCK_N   64
#define STAGE_DW  32
#define PAD       36

#if __has_builtin(__builtin_amdgcn_global_load_async_to_lds_b128) && \
    __has_builtin(__builtin_amdgcn_s_wait_asynccnt)
#define FSU_ASYNC 1
#else
#define FSU_ASYNC 0
#endif

__global__ __launch_bounds__(256) void pack_x_kernel(
    const float* __restrict__ x, uint32_t* __restrict__ x8,
    float* __restrict__ rowsum, int IN)
{
  __shared__ int sdata[256];
  const int b = blockIdx.x;
  const int t = threadIdx.x;
  const int nvec = IN >> 2;
  int local = 0;
  for (int d = t; d < nvec; d += 256) {
    const float4 v = ((const float4*)(x + (size_t)b * IN))[d];
    const int b0 = v.x > 0.5f, b1 = v.y > 0.5f, b2 = v.z > 0.5f, b3 = v.w > 0.5f;
    x8[(size_t)b * nvec + d] =
        (uint32_t)b0 | ((uint32_t)b1 << 8) | ((uint32_t)b2 << 16) | ((uint32_t)b3 << 24);
    local += b0 + b1 + b2 + b3;
  }
  sdata[t] = local;
  __syncthreads();
  for (int s = 128; s > 0; s >>= 1) {
    if (t < s) sdata[t] += sdata[t + s];
    __syncthreads();
  }
  if (t == 0) rowsum[b] = (float)sdata[0];
}

__global__ __launch_bounds__(256) void pack_w_kernel(
    const float* __restrict__ weight_bin, const int* __restrict__ wrdx,
    const float* __restrict__ rng, const float* __restrict__ bias_bin,
    const int* __restrict__ brdx, const float* __restrict__ brng,
    uint32_t* __restrict__ w8, float* __restrict__ colsum,
    float* __restrict__ bbit, int IN, int n_rng, int n_brng)
{
  __shared__ int sdata[256];
  const int o = blockIdx.x;
  const int t = threadIdx.x;
  const int nvec = IN >> 2;
  int local = 0;
  for (int d = t; d < nvec; d += 256) {
    const size_t base = (size_t)o * IN + (size_t)d * 4;
    uint32_t p = 0;
    int s = 0;
#pragma unroll
    for (int j = 0; j < 4; ++j) {
      const float wb = weight_bin[base + j];
      const unsigned rdx = (unsigned)wrdx[base + j] % (unsigned)n_rng;
      const int bit = wb > rng[rdx];
      p |= (uint32_t)bit << (8 * j);
      s += bit;
    }
    w8[(size_t)o * nvec + d] = p;
    local += s;
  }
  sdata[t] = local;
  __syncthreads();
  for (int s = 128; s > 0; s >>= 1) {
    if (t < s) sdata[t] += sdata[t + s];
    __syncthreads();
  }
  if (t == 0) {
    colsum[o] = (float)sdata[0];
    const unsigned rdx = (unsigned)brdx[o] % (unsigned)n_brng;
    bbit[o] = (bias_bin[o] > brng[rdx]) ? 1.0f : 0.0f;
  }
}

__device__ inline v8i load_a_frag(const uint32_t* arow, int h) {
  const int2 a0 = *(const int2*)(arow + 2 * h);
  const int2 a1 = *(const int2*)(arow + 4 + 2 * h);
  const int2 a2 = *(const int2*)(arow + 8 + 2 * h);
  const int2 a3 = *(const int2*)(arow + 12 + 2 * h);
  v8i A;
  A[0] = a0.x; A[1] = a0.y; A[2] = a1.x; A[3] = a1.y;
  A[4] = a2.x; A[5] = a2.y; A[6] = a3.x; A[7] = a3.y;
  return A;
}
__device__ inline v8i load_b_frag(const uint32_t* brow, int h) {
  const int2 b0 = *(const int2*)(brow + 2 * h);
  const int2 b1 = *(const int2*)(brow + 4 + 2 * h);
  const int2 b2 = *(const int2*)(brow + 8 + 2 * h);
  const int2 b3 = *(const int2*)(brow + 12 + 2 * h);
  v8i B;
  B[0] = b0.x; B[1] = b0.y; B[2] = b1.x; B[3] = b1.y;
  B[4] = b2.x; B[5] = b2.y; B[6] = b3.x; B[7] = b3.y;
  return B;
}

__global__ __launch_bounds__(256, 1) void fsulinear_wmma_kernel(
    const uint32_t* __restrict__ x8, const uint32_t* __restrict__ w8,
    const float* __restrict__ rowsum, const float* __restrict__ colsum,
    const float* __restrict__ bbit, const int* __restrict__ scale_p,
    float* __restrict__ out, int IN, int OUT)
{
  __shared__ uint32_t lds_a[2][BLOCK_M * PAD];
  __shared__ uint32_t lds_b[2][BLOCK_N * PAD];

  const int t    = threadIdx.x;
  const int lane = t & 31;
  const int wave = t >> 5;
  const int h    = lane >> 4;
  const int l16  = lane & 15;
  const int kvec = IN >> 2;
  const int m_blk = blockIdx.x * BLOCK_M;
  const int n_blk = blockIdx.y * BLOCK_N;
  const int nstages = kvec / STAGE_DW;

  v8i acc0 = {}, acc1 = {}, acc2 = {}, acc3 = {};

  int a_r[4], a_c[4], b_r[2], b_c[2];
#pragma unroll
  for (int p = 0; p < 4; ++p) { const int idx = p * 1024 + t * 4; a_r[p] = idx >> 5; a_c[p] = idx & 31; }
#pragma unroll
  for (int p = 0; p < 2; ++p) { const int idx = p * 1024 + t * 4; b_r[p] = idx >> 5; b_c[p] = idx & 31; }

  auto consume = [&](int buf) {
    const uint32_t* abase = &lds_a[buf][(wave * 16 + l16) * PAD];
    const uint32_t* b0 = &lds_b[buf][( 0 + l16) * PAD];
    const uint32_t* b1 = &lds_b[buf][(16 + l16) * PAD];
    const uint32_t* b2 = &lds_b[buf][(32 + l16) * PAD];
    const uint32_t* b3 = &lds_b[buf][(48 + l16) * PAD];
#pragma unroll
    for (int ks = 0; ks < STAGE_DW; ks += 16) {
      const v8i A = load_a_frag(abase + ks, h);
      acc0 = __builtin_amdgcn_wmma_i32_16x16x64_iu8(false, A, false, load_b_frag(b0 + ks, h), acc0, false, false);
      acc1 = __builtin_amdgcn_wmma_i32_16x16x64_iu8(false, A, false, load_b_frag(b1 + ks, h), acc1, false, false);
      acc2 = __builtin_amdgcn_wmma_i32_16x16x64_iu8(false, A, false, load_b_frag(b2 + ks, h), acc2, false, false);
      acc3 = __builtin_amdgcn_wmma_i32_16x16x64_iu8(false, A, false, load_b_frag(b3 + ks, h), acc3, false, false);
    }
  };

#if FSU_ASYNC
  auto stage_async = [&](int s, int buf) {
#pragma unroll
    for (int p = 0; p < 4; ++p)
      __builtin_amdgcn_global_load_async_to_lds_b128(
          (g4i_as1*)(x8 + (size_t)(m_blk + a_r[p]) * kvec + s * STAGE_DW + a_c[p]),
          (g4i_as3*)&lds_a[buf][a_r[p] * PAD + a_c[p]],
          0, 0);
#pragma unroll
    for (int p = 0; p < 2; ++p)
      __builtin_amdgcn_global_load_async_to_lds_b128(
          (g4i_as1*)(w8 + (size_t)(n_blk + b_r[p]) * kvec + s * STAGE_DW + b_c[p]),
          (g4i_as3*)&lds_b[buf][b_r[p] * PAD + b_c[p]],
          0, 0);
  };

  stage_async(0, 0);
  for (int s = 0; s < nstages - 1; ++s) {
    stage_async(s + 1, (s + 1) & 1);
    __builtin_amdgcn_s_wait_asynccnt(6);
    __syncthreads();
    consume(s & 1);
    __syncthreads();
  }
  __builtin_amdgcn_s_wait_asynccnt(0);
  __syncthreads();
  consume((nstages - 1) & 1);
#else
  uint4 ar[4], br[2];
  auto g_load = [&](int s) {
#pragma unroll
    for (int p = 0; p < 4; ++p)
      ar[p] = *(const uint4*)(x8 + (size_t)(m_blk + a_r[p]) * kvec + s * STAGE_DW + a_c[p]);
#pragma unroll
    for (int p = 0; p < 2; ++p)
      br[p] = *(const uint4*)(w8 + (size_t)(n_blk + b_r[p]) * kvec + s * STAGE_DW + b_c[p]);
  };
  auto l_store = [&](int buf) {
#pragma unroll
    for (int p = 0; p < 4; ++p)
      *(uint4*)(&lds_a[buf][a_r[p] * PAD + a_c[p]]) = ar[p];
#pragma unroll
    for (int p = 0; p < 2; ++p)
      *(uint4*)(&lds_b[buf][b_r[p] * PAD + b_c[p]]) = br[p];
  };
  g_load(0);
  l_store(0);
  for (int s = 0; s < nstages; ++s) {
    __syncthreads();
    if (s + 1 < nstages) g_load(s + 1);
    consume(s & 1);
    if (s + 1 < nstages) l_store((s + 1) & 1);
  }
#endif

  const float scale_f = (float)(*scale_p);
  const float offset  = ((float)IN + 1.0f - scale_f) * 0.5f;
  const float acc_max = (float)(1 << (FSU_DEPTH - 2));
  const float basec   = (float)IN;
  const int   m_base  = m_blk + wave * 16;

  float rs[8];
#pragma unroll
  for (int r = 0; r < 8; ++r) rs[r] = rowsum[m_base + r + 8 * h];

  const v8i accs[4] = {acc0, acc1, acc2, acc3};
#pragma unroll
  for (int tt = 0; tt < 4; ++tt) {
    const int n = n_blk + tt * 16 + l16;
    const float cn = bbit[n] - colsum[n];
#pragma unroll
    for (int r = 0; r < 8; ++r) {
      const int m = m_base + r + 8 * h;
      const float pc = basec + cn - rs[r] + 2.0f * (float)accs[tt][r];
      float a = pc - offset;
      a = fminf(fmaxf(a, -acc_max), acc_max);
      out[(size_t)m * OUT + n] = (a >= scale_f) ? 1.0f : 0.0f;
    }
  }
}

extern "C" void kernel_launch(void* const* d_in, const int* in_sizes, int n_in,
                              void* d_out, int out_size, void* d_ws, size_t ws_size,
                              hipStream_t stream) {
  (void)n_in; (void)out_size; (void)ws_size;
  const float* x          = (const float*)d_in[0];
  const float* weight_bin = (const float*)d_in[1];
  const float* bias_bin   = (const float*)d_in[2];
  const float* rng        = (const float*)d_in[3];
  const float* brng       = (const float*)d_in[4];
  const int*   wrdx       = (const int*)d_in[5];
  const int*   brdx       = (const int*)d_in[6];
  const int*   scale      = (const int*)d_in[7];

  const int OUT   = in_sizes[2];
  const int IN    = in_sizes[1] / OUT;
  const int Bn    = in_sizes[0] / IN;
  const int nrng  = in_sizes[3];
  const int nbrng = in_sizes[4];

  uint8_t* ws = (uint8_t*)d_ws;
  size_t off = 0;
  uint32_t* x8     = (uint32_t*)(ws + off); off += (size_t)Bn * IN;
  uint32_t* w8     = (uint32_t*)(ws + off); off += (size_t)OUT * IN;
  float*    rowsum = (float*)(ws + off);    off += (size_t)Bn * sizeof(float);
  float*    colsum = (float*)(ws + off);    off += (size_t)OUT * sizeof(float);
  float*    bbitp  = (float*)(ws + off);

  pack_x_kernel<<<Bn, 256, 0, stream>>>(x, x8, rowsum, IN);
  pack_w_kernel<<<OUT, 256, 0, stream>>>(weight_bin, wrdx, rng, bias_bin, brdx, brng,
                                         w8, colsum, bbitp, IN, nrng, nbrng);

  dim3 grid(Bn / BLOCK_M, OUT / BLOCK_N);
  fsulinear_wmma_kernel<<<grid, 256, 0, stream>>>(x8, w8, rowsum, colsum, bbitp,
                                                  scale, (float*)d_out, IN, OUT);
}
